// SE3Transformer_16698832847173
// MI455X (gfx1250) — matrix operand order repaired and checked by reading, not compiled, not run on hardware
//
#include <hip/hip_runtime.h>

// ---------------------------------------------------------------------------
// SE(3)-Transformer forward for MI455X (gfx1250, wave32, WMMA).
// Heavy matmuls (radial w3, cross-attention QK^T and AV) use
// v_wmma_f32_16x16x32_f16. The radial GEMM loads both fragments as packed
// 256-bit f16 vectors (weights pre-packed per-tile, activations stored in
// fragment K-order by the producer kernel).
// ---------------------------------------------------------------------------

#define EHA 8192
#define ECA 4096
#define NHA 2048
#define NCA 1024
#define L0HA 82
#define L0CA 60
#define CHUNK 2048

typedef __attribute__((ext_vector_type(16))) _Float16 v16h;
typedef __attribute__((ext_vector_type(8)))  float    v8f;

__device__ __forceinline__ v8f wmma32(v16h a, v16h b, v8f c) {
  return __builtin_amdgcn_wmma_f32_16x16x32_f16(false, a, false, b, (short)0, c, false, false);
}

// ---------------------------- geometry / basis -----------------------------
__global__ void k_edge_geom(const float* __restrict__ w, const float* __restrict__ d,
                            float* __restrict__ feat, float* __restrict__ y1,
                            float* __restrict__ b11, int E) {
  int e = blockIdx.x * blockDim.x + threadIdx.x;
  if (e >= E) return;
  float dx = d[3*e], dy = d[3*e+1], dz = d[3*e+2];
  float r  = sqrtf(dx*dx + dy*dy + dz*dz);
  float rc = fmaxf(r, 1e-8f);
  float x = dx/rc, y = dy/rc, z = dz/rc;
#pragma unroll
  for (int i = 0; i < 5; i++) feat[6*e+i] = w[5*e+i];
  feat[6*e+5] = r;
  const float c1 = 0.4886025119029199f;
  float Y1[3] = {c1*y, c1*z, c1*x};
  y1[3*e] = Y1[0]; y1[3*e+1] = Y1[1]; y1[3*e+2] = Y1[2];
  float Y2[5] = {1.0925484305920792f*x*y, 1.0925484305920792f*y*z,
                 0.31539156525252005f*(3.f*z*z - 1.f), 1.0925484305920792f*x*z,
                 0.5462742152960396f*(x*x - y*y)};
  const float is2 = 0.70710678118654752f, is6 = 0.40824829046386302f;
  const float d0  = 0.57735026918962576f * 0.28209479177387814f;
  float B[3][3][3];
#pragma unroll
  for (int p = 0; p < 3; p++)
#pragma unroll
    for (int q = 0; q < 3; q++) { B[p][q][0] = (p==q)?d0:0.f; B[p][q][1]=0.f; B[p][q][2]=0.f; }
  // f=1 : eps/sqrt(2) . Y1
  B[0][1][1] =  is2*Y1[2]; B[1][0][1] = -is2*Y1[2];
  B[0][2][1] = -is2*Y1[1]; B[2][0][1] =  is2*Y1[1];
  B[1][2][1] =  is2*Y1[0]; B[2][1][1] = -is2*Y1[0];
  // f=2 : permuted symmetric basis . Y2
  B[0][0][2] = -is6*Y2[2] - is2*Y2[4];
  B[1][1][2] =  2.f*is6*Y2[2];
  B[2][2][2] = -is6*Y2[2] + is2*Y2[4];
  B[0][1][2] = B[1][0][2] = is2*Y2[1];
  B[1][2][2] = B[2][1][2] = is2*Y2[3];
  B[0][2][2] = B[2][0][2] = is2*Y2[0];
#pragma unroll
  for (int p = 0; p < 3; p++)
#pragma unroll
    for (int q = 0; q < 3; q++)
#pragma unroll
      for (int f = 0; f < 3; f++)
        b11[(size_t)27*e + (p*3+q)*3 + f] = B[p][q][f];
}

// ------------------------- radial net front (2 LN-MLP layers) --------------
// Output row stored in WMMA A-fragment K-order: [K0..7, K16..23, K8..15, K24..31]
__device__ __forceinline__ int kpos(int j) {
  return (j < 8 || j >= 24) ? j : ((j < 16) ? j + 8 : j - 8);
}
__global__ void k_rad_front(const float* __restrict__ feat, int E,
                            const float* w1, const float* b1, const float* g1, const float* bb1,
                            const float* w2, const float* b2, const float* g2, const float* bb2,
                            _Float16* __restrict__ h2out) {
  __shared__ float sw1[6*32], sw2[32*32];
  __shared__ float sb1[32], sg1[32], sbb1[32], sb2[32], sg2[32], sbb2[32];
  for (int i = threadIdx.x; i < 192;  i += blockDim.x) sw1[i] = w1[i];
  for (int i = threadIdx.x; i < 1024; i += blockDim.x) sw2[i] = w2[i];
  if (threadIdx.x < 32) {
    int c = threadIdx.x;
    sb1[c]=b1[c]; sg1[c]=g1[c]; sbb1[c]=bb1[c];
    sb2[c]=b2[c]; sg2[c]=g2[c]; sbb2[c]=bb2[c];
  }
  __syncthreads();
  int e = blockIdx.x * blockDim.x + threadIdx.x;
  if (e >= E) return;
  float f[6];
#pragma unroll
  for (int i = 0; i < 6; i++) f[i] = feat[6*e+i];
  float h[32];
#pragma unroll
  for (int j = 0; j < 32; j++) {
    float a = sb1[j];
#pragma unroll
    for (int i = 0; i < 6; i++) a += f[i]*sw1[i*32+j];
    h[j] = a;
  }
  float mu = 0.f;
#pragma unroll
  for (int j = 0; j < 32; j++) mu += h[j];
  mu *= (1.f/32.f);
  float var = 0.f;
#pragma unroll
  for (int j = 0; j < 32; j++) { float t = h[j]-mu; var += t*t; }
  var *= (1.f/32.f);
  float rs = rsqrtf(var + 1e-5f);
#pragma unroll
  for (int j = 0; j < 32; j++) h[j] = fmaxf(sg1[j]*(h[j]-mu)*rs + sbb1[j], 0.f);
  float h2[32];
#pragma unroll
  for (int j = 0; j < 32; j++) {
    float a = sb2[j];
    for (int i = 0; i < 32; i++) a += h[i]*sw2[i*32+j];
    h2[j] = a;
  }
  mu = 0.f;
#pragma unroll
  for (int j = 0; j < 32; j++) mu += h2[j];
  mu *= (1.f/32.f);
  var = 0.f;
#pragma unroll
  for (int j = 0; j < 32; j++) { float t = h2[j]-mu; var += t*t; }
  var *= (1.f/32.f);
  rs = rsqrtf(var + 1e-5f);
#pragma unroll
  for (int j = 0; j < 32; j++)
    h2out[(size_t)e*32 + kpos(j)] = (_Float16)fmaxf(sg2[j]*(h2[j]-mu)*rs + sbb2[j], 0.f);
}

// ---- pack W[32,N] f32 into per-tile WMMA B fragments (f16, 16 halves/lane) --
__global__ void k_pack_w3(const float* __restrict__ W, _Float16* __restrict__ Wp, int N) {
  int idx = blockIdx.x*blockDim.x + threadIdx.x;       // ntiles*32*16 threads
  int total = (N/16)*32*16;
  if (idx >= total) return;
  int c    = idx & 15;
  int lane = (idx >> 4) & 31;
  int nt   = idx >> 9;
  int kb2  = (lane < 16) ? 0 : 8;
  int n    = nt*16 + (lane & 15);
  Wp[idx] = (_Float16)W[(size_t)(kb2 + c + ((c >= 8) ? 8 : 0))*N + n];
}

// ---------------- WMMA GEMM: out[M,N] = A_f16[M,32] @ W[32,N] + bias --------
// A rows stored in fragment K-order; Wp pre-packed per tile.
__global__ void k_gemm_h2w3(const _Float16* __restrict__ A, const _Float16* __restrict__ Wp,
                            const float* __restrict__ bias, float* __restrict__ out,
                            int M, int N) {
  int lane = threadIdx.x;            // 32 threads = 1 wave
  int mt = blockIdx.x;               // M/16
  int ng = blockIdx.y;               // groups of 4 N-tiles
  int mrow = mt*16 + (lane & 15);
  v16h a = *(const v16h*)(A + (size_t)mrow*32 + ((lane < 16) ? 0 : 16));
  int mb = mt*16 + ((lane < 16) ? 0 : 8);
#pragma unroll
  for (int nt = 0; nt < 4; nt++) {
    int tile = ng*4 + nt;
    int n = tile*16 + (lane & 15);
    v16h b = *(const v16h*)(Wp + ((size_t)tile*32 + lane)*16);
    v8f c = {};
    c = wmma32(a, b, c);
    float bv = bias[n];
#pragma unroll
    for (int r = 0; r < 8; r++) out[(size_t)(mb + r)*N + n] = c[r] + bv;
  }
}

// -------- pair_msg: out[e,m,p] += sum_{n,q,f} R[e,m,n,f] B[e,p,q,f] h[src,n,q]
__global__ void k_pair_msg(const float* __restrict__ R,
                           const float* __restrict__ y1, const float* __restrict__ b11,
                           const float* __restrict__ hsrc, const int* __restrict__ src,
                           float* __restrict__ out, int di, int dop, int mo, int ebase) {
  __shared__ float hs[192];
  __shared__ float G[576];
  __shared__ float ps[192];
  int ce = blockIdx.x;
  int e  = ebase + ce;
  int t  = threadIdx.x;              // 64 threads
  int si = src[e];
  if (di == 0) {
    hs[t] = hsrc[(size_t)si*64 + t];
  } else {
#pragma unroll
    for (int j = 0; j < 3; j++) hs[t*3+j] = hsrc[((size_t)si*64 + t)*3 + j];
  }
  __syncthreads();
  if (di == 1) {
    if (dop == 1) {
#pragma unroll
      for (int p = 0; p < 3; p++)
#pragma unroll
        for (int f = 0; f < 3; f++) {
          float g = 0.f;
#pragma unroll
          for (int q = 0; q < 3; q++) g += b11[(size_t)27*e + (p*3+q)*3 + f]*hs[t*3+q];
          G[(p*64 + t)*3 + f] = g;
        }
    } else {
      float g = 0.f;
#pragma unroll
      for (int q = 0; q < 3; q++) g += y1[3*e+q]*hs[t*3+q];
      G[t] = g;
    }
  }
  __syncthreads();
  int m = t % mo, grp = t / mo, ngrp = 64 / mo;
  int n0 = grp * mo;
  float acc0 = 0.f, acc1 = 0.f, acc2 = 0.f;
  int nf = (di && dop) ? 3 : 1;
  const float* Re = R + (size_t)ce * mo * 64 * nf;
  if (di == 0 && dop == 0) {
    for (int n = n0; n < n0+mo; n++) acc0 += Re[m*64+n]*hs[n];
    acc0 *= 0.28209479177387814f;
  } else if (di == 0 && dop == 1) {
    float t0 = 0.f;
    for (int n = n0; n < n0+mo; n++) t0 += Re[m*64+n]*hs[n];
    acc0 = t0*y1[3*e]; acc1 = t0*y1[3*e+1]; acc2 = t0*y1[3*e+2];
  } else if (di == 1 && dop == 0) {
    for (int n = n0; n < n0+mo; n++) acc0 += Re[m*64+n]*G[n];
  } else {
    for (int n = n0; n < n0+mo; n++)
#pragma unroll
      for (int f = 0; f < 3; f++) {
        float rv = Re[(m*64+n)*3 + f];
        acc0 += rv*G[(0*64+n)*3+f];
        acc1 += rv*G[(1*64+n)*3+f];
        acc2 += rv*G[(2*64+n)*3+f];
      }
  }
  ps[t*3] = acc0; ps[t*3+1] = acc1; ps[t*3+2] = acc2;
  __syncthreads();
  for (int st = ngrp >> 1; st > 0; st >>= 1) {
    if (grp < st) {
      int o2 = t + st*mo;
      ps[t*3]   += ps[o2*3];
      ps[t*3+1] += ps[o2*3+1];
      ps[t*3+2] += ps[o2*3+2];
    }
    __syncthreads();
  }
  int P = 2*dop + 1;
  if (t < mo) {
    for (int p = 0; p < P; p++)
      out[((size_t)e*mo + m)*P + p] += ps[t*3+p];
  }
}

// -------------------- g1x1 (with optional concat, strided out) -------------
__global__ void k_g1x1(const float* __restrict__ W, const float* __restrict__ A, int Ca,
                       const float* __restrict__ B, int Cb, float* __restrict__ out,
                       int N, int O, int M, int outStride, int outOff) {
  int idx = blockIdx.x*blockDim.x + threadIdx.x;
  if (idx >= N*O) return;
  int n = idx / O, o = idx % O;
  int C = Ca + Cb;
  for (int m = 0; m < M; m++) {
    float acc = 0.f;
    for (int c = 0; c < Ca; c++) acc += W[o*C + c] * A[((size_t)n*Ca + c)*M + m];
    for (int c = 0; c < Cb; c++) acc += W[o*C + Ca + c] * B[((size_t)n*Cb + c)*M + m];
    out[((size_t)n*outStride + outOff + o)*M + m] = acc;
  }
}

// ------------------------------ dense linear --------------------------------
__global__ void k_linear(const float* __restrict__ in, const float* __restrict__ W,
                         const float* __restrict__ b, float* __restrict__ out,
                         int N, int I, int O, int act) {
  int idx = blockIdx.x*blockDim.x + threadIdx.x;
  if (idx >= N*O) return;
  int n = idx / O, o = idx % O;
  float acc = b[o];
  for (int i = 0; i < I; i++) acc += in[(size_t)n*I + i] * W[(size_t)i*O + o];
  if (act) acc = (acc > 0.f) ? acc : expm1f(acc);
  out[(size_t)n*O + o] = acc;
}

// --------------------------- edge attention ---------------------------------
__device__ __forceinline__ unsigned enc_f(float f) {
  unsigned u = __float_as_uint(f);
  return (u & 0x80000000u) ? ~u : (u | 0x80000000u);
}
__device__ __forceinline__ float dec_f(unsigned u) {
  return (u & 0x80000000u) ? __uint_as_float(u ^ 0x80000000u) : __uint_as_float(~u);
}

__global__ void k_att_score(const float* k0, const float* k1, const float* q0, const float* q1,
                            const int* dst, float* s, unsigned* mbuf, int E, int full, float scale) {
  int idx = blockIdx.x*blockDim.x + threadIdx.x;
  if (idx >= E*4) return;
  int e = idx >> 2, h = idx & 3;
  int dn = dst[e];
  float acc = 0.f;
#pragma unroll
  for (int c = 0; c < 4; c++) acc += k0[(size_t)e*16 + h*4 + c] * q0[(size_t)dn*16 + h*4 + c];
  if (full)
#pragma unroll
    for (int j = 0; j < 12; j++) acc += k1[(size_t)e*48 + h*12 + j] * q1[(size_t)dn*48 + h*12 + j];
  acc *= scale;
  s[idx] = acc;
  atomicMax(&mbuf[dn*4 + h], enc_f(acc));
}

__global__ void k_att_exp(float* s, const unsigned* mbuf, float* ssum, const int* dst, int E) {
  int idx = blockIdx.x*blockDim.x + threadIdx.x;
  if (idx >= E*4) return;
  int e = idx >> 2, h = idx & 3;
  int dn = dst[e];
  float ex = expf(s[idx] - dec_f(mbuf[dn*4 + h]));
  s[idx] = ex;
  atomicAdd(&ssum[dn*4 + h], ex);
}

__global__ void k_att_scatter(const float* s, const float* ssum, const int* dst,
                              const float* v0, const float* v1,
                              float* z0, float* z1, int E) {
  int idx = blockIdx.x*blockDim.x + threadIdx.x;
  if (idx >= E*4) return;
  int e = idx >> 2, h = idx & 3;
  int dn = dst[e];
  float a = s[idx] / (ssum[dn*4 + h] + 1e-9f);
#pragma unroll
  for (int c = 0; c < 4; c++)
    atomicAdd(&z0[(size_t)dn*16 + h*4 + c], a * v0[(size_t)e*16 + h*4 + c]);
#pragma unroll
  for (int j = 0; j < 12; j++)
    atomicAdd(&z1[(size_t)dn*48 + h*12 + j], a * v1[(size_t)e*48 + h*12 + j]);
}

// ------------------------------- gnorm --------------------------------------
__global__ void k_gnorm(float* __restrict__ x, const float* g, const float* b, int M) {
  int n = blockIdx.x, c = threadIdx.x;  // 64 threads
  __shared__ float sh[64];
  float xv[3];
  float ss = 0.f;
  for (int m = 0; m < M; m++) { xv[m] = x[((size_t)n*64 + c)*M + m]; ss += xv[m]*xv[m]; }
  float nrm = sqrtf(fmaxf(ss, 1e-12f));
  sh[c] = nrm;
  __syncthreads();
  float mu = 0.f;
  for (int i = 0; i < 64; i++) mu += sh[i];
  mu *= (1.f/64.f);
  float var = 0.f;
  for (int i = 0; i < 64; i++) { float t = sh[i]-mu; var += t*t; }
  var *= (1.f/64.f);
  float scale = fmaxf(g[c]*(nrm-mu)*rsqrtf(var + 1e-5f) + b[c], 0.f);
  float inv = scale / nrm;
  for (int m = 0; m < M; m++) x[((size_t)n*64 + c)*M + m] = xv[m]*inv;
}

// ------------------- cross attention: S = scale * Q K^T (WMMA) --------------
__global__ void k_qk(const float* __restrict__ Q, int qStride,
                     const float* __restrict__ Kf, int kStride,
                     float* __restrict__ S, int N, int Kdim, float scale) {
  int lane = threadIdx.x;
  int mt = blockIdx.x, nt = blockIdx.y;
  int KT = (Kdim + 31) / 32;
  v8f c = {};
  int mrow = mt*16 + (lane & 15);
  int kbA = (lane < 16) ? 0 : 8;
  int ncol = nt*16 + (lane & 15);
  int kbB = (lane < 16) ? 0 : 8;
  for (int kt = 0; kt < KT; kt++) {
    union { v16h v; _Float16 h[16]; } a, b;
#pragma unroll
    for (int i = 0; i < 8; i++) {
      int K = kt*32 + ((i < 4) ? (kbA + 2*i) : (16 + kbA + 2*(i-4)));
      a.h[2*i]   = (K   < Kdim) ? (_Float16)Q[(size_t)mrow*qStride + K]     : (_Float16)0.f;
      a.h[2*i+1] = (K+1 < Kdim) ? (_Float16)Q[(size_t)mrow*qStride + K + 1] : (_Float16)0.f;
    }
#pragma unroll
    for (int i = 0; i < 8; i++) {
      int K = kt*32 + ((i < 4) ? (kbB + 2*i) : (16 + kbB + 2*(i-4)));
      b.h[2*i]   = (K   < Kdim) ? (_Float16)Kf[(size_t)ncol*kStride + K]     : (_Float16)0.f;
      b.h[2*i+1] = (K+1 < Kdim) ? (_Float16)Kf[(size_t)ncol*kStride + K + 1] : (_Float16)0.f;
    }
    c = wmma32(a.v, b.v, c);
  }
  int mb = mt*16 + ((lane < 16) ? 0 : 8);
#pragma unroll
  for (int r = 0; r < 8; r++) S[(size_t)(mb + r)*N + ncol] = c[r]*scale;
}

// ------------------------------ row softmax ---------------------------------
__global__ void k_softmax_row(float* __restrict__ S, int N) {
  int row = blockIdx.x, tid = threadIdx.x;
  float* p = S + (size_t)row*N;
  __shared__ float red[256];
  float m = -1e30f;
  for (int i = tid; i < N; i += 256) m = fmaxf(m, p[i]);
  red[tid] = m; __syncthreads();
  for (int s = 128; s > 0; s >>= 1) { if (tid < s) red[tid] = fmaxf(red[tid], red[tid+s]); __syncthreads(); }
  m = red[0]; __syncthreads();
  float sum = 0.f;
  for (int i = tid; i < N; i += 256) { float ex = expf(p[i]-m); p[i] = ex; sum += ex; }
  red[tid] = sum; __syncthreads();
  for (int s = 128; s > 0; s >>= 1) { if (tid < s) red[tid] += red[tid+s]; __syncthreads(); }
  float inv = 1.f / red[0];
  for (int i = tid; i < N; i += 256) p[i] *= inv;
}

// ----------------- out[:, off+ch, mm] = A[M,Ncols] @ V (WMMA) ---------------
__global__ void k_av(const float* __restrict__ A, int Ncols,
                     const float* __restrict__ V, int vStride, int vOff,
                     float* __restrict__ out, int outStride, int outOff,
                     int M3, int Cdim) {
  int lane = threadIdx.x;
  int mt = blockIdx.x, nt = blockIdx.y;
  v8f c = {};
  int mrow = mt*16 + (lane & 15);
  int kbA = (lane < 16) ? 0 : 8;
  int ccol = nt*16 + (lane & 15);
  int kbB = (lane < 16) ? 0 : 8;
  bool cok = ccol < Cdim;
  for (int kt = 0; kt < Ncols/32; kt++) {
    union { v16h v; _Float16 h[16]; } a, b;
#pragma unroll
    for (int i = 0; i < 8; i++) {
      int K = kt*32 + ((i < 4) ? (kbA + 2*i) : (16 + kbA + 2*(i-4)));
      a.h[2*i]   = (_Float16)A[(size_t)mrow*Ncols + K];
      a.h[2*i+1] = (_Float16)A[(size_t)mrow*Ncols + K + 1];
    }
#pragma unroll
    for (int i = 0; i < 8; i++) {
      int K = kt*32 + ((i < 4) ? (kbB + 2*i) : (16 + kbB + 2*(i-4)));
      b.h[2*i]   = cok ? (_Float16)V[(size_t)K*vStride + vOff + ccol]     : (_Float16)0.f;
      b.h[2*i+1] = cok ? (_Float16)V[(size_t)(K+1)*vStride + vOff + ccol] : (_Float16)0.f;
    }
    c = wmma32(a.v, b.v, c);
  }
  if (!cok) return;
  int ch = ccol / M3, mm = ccol % M3;
  int mb = mt*16 + ((lane < 16) ? 0 : 8);
#pragma unroll
  for (int r = 0; r < 8; r++)
    out[((size_t)(mb + r)*outStride + outOff + ch)*M3 + mm] = c[r];
}

// ------------------------------- final conv ---------------------------------
__global__ void k_final_scatter(const float* __restrict__ msg, const float* __restrict__ selfW,
                                const float* __restrict__ h0, const int* __restrict__ dst,
                                float* acc, float* deg, int E) {
  int e = blockIdx.x*blockDim.x + threadIdx.x;
  if (e >= E) return;
  int dn = dst[e];
  float sd = 0.f;
  for (int c = 0; c < 64; c++) sd += selfW[c]*h0[(size_t)dn*64 + c];
  atomicAdd(&acc[dn], msg[e] + sd);
  atomicAdd(&deg[dn], 1.f);
}
__global__ void k_final_div(const float* acc, const float* deg, float* out, int N) {
  int n = blockIdx.x*blockDim.x + threadIdx.x;
  if (n >= N) return;
  out[n] = acc[n] / fmaxf(deg[n], 1.f);
}

// ===========================================================================
// Host side: param pytree decode (jax sorted-dict flatten order) + schedule
// ===========================================================================
struct RadP { const float *b1,*b2,*b3,*bb1,*bb2,*g1,*g2,*w1,*w2,*w3; };
struct LayerP {
  bool full;
  RadP k00,k01,k10,k11;
  const float *mid0,*mid1,*nb0,*ng0,*nb1,*ng1,*proj0,*proj1,*q0,*q1;
  RadP v00,v01,v10,v11;
};
struct GraphP {
  RadP fin00, fin10; const float* fin_self;
  LayerP L[4];
  const float *lin1_b,*lin1_w,*lin2_b,*lin2_w;
};

static RadP takeRad(void* const* in, int& i) {
  RadP r;
  r.b1 =(const float*)in[i++]; r.b2 =(const float*)in[i++]; r.b3 =(const float*)in[i++];
  r.bb1=(const float*)in[i++]; r.bb2=(const float*)in[i++];
  r.g1 =(const float*)in[i++]; r.g2 =(const float*)in[i++];
  r.w1 =(const float*)in[i++]; r.w2 =(const float*)in[i++]; r.w3 =(const float*)in[i++];
  return r;
}
static GraphP takeGraph(void* const* in, int& i) {
  GraphP g;
  g.fin00 = takeRad(in, i); g.fin10 = takeRad(in, i);
  g.fin_self = (const float*)in[i++];
  for (int l = 0; l < 4; l++) {
    LayerP& L = g.L[l]; L.full = (l > 0);
    L.k00 = takeRad(in, i);
    if (L.full) { L.k01 = takeRad(in, i); L.k10 = takeRad(in, i); L.k11 = takeRad(in, i); }
    L.mid0 = (const float*)in[i++]; L.mid1 = (const float*)in[i++];
    L.nb0 = (const float*)in[i++]; L.ng0 = (const float*)in[i++];
    L.nb1 = (const float*)in[i++]; L.ng1 = (const float*)in[i++];
    L.proj0 = (const float*)in[i++]; L.proj1 = (const float*)in[i++];
    L.q0 = (const float*)in[i++];
    if (L.full) L.q1 = (const float*)in[i++]; else L.q1 = L.q0;
    L.v00 = takeRad(in, i); L.v01 = takeRad(in, i);
    if (L.full) { L.v10 = takeRad(in, i); L.v11 = takeRad(in, i); }
  }
  g.lin1_b = (const float*)in[i++]; g.lin1_w = (const float*)in[i++];
  g.lin2_b = (const float*)in[i++]; g.lin2_w = (const float*)in[i++];
  return g;
}

struct Graph {
  int N, E, L0, mid2;
  const float* f0; const int *src, *dst;
  float *feat, *y1, *b11;
  float *h0[2], *h1[2];
};
struct Scratch {
  float *tmp0,*tmp1,*embt,*Rbuf,*v0,*v1,*k0,*k1,*q0,*q1,*s,*ssum,*z0,*z1,*msg,*acc,*deg;
  unsigned* mbuf;
  _Float16 *h2, *wpack;
};

struct Bump { char* p; size_t off; };
static void* bget(Bump& b, size_t bytes) {
  size_t o = (b.off + 255) & ~(size_t)255;
  b.off = o + bytes;
  return b.p + o;
}
static int cdiv(int a, int b) { return (a + b - 1) / b; }

static void run_rad(const Graph& g, Scratch& S, const RadP& r, int di, int dop, int mo,
                    float* outbuf, int cur, hipStream_t st) {
  int nf = (di && dop) ? 3 : 1;
  int Ncols = nf * mo * 64;
  k_rad_front<<<cdiv(g.E,128), 128, 0, st>>>(g.feat, g.E, r.w1, r.b1, r.g1, r.bb1,
                                             r.w2, r.b2, r.g2, r.bb2, S.h2);
  k_pack_w3<<<cdiv((Ncols/16)*512,256), 256, 0, st>>>(r.w3, S.wpack, Ncols);
  const float* hsrc = di ? g.h1[cur] : g.h0[cur];
  for (int cb = 0; cb < g.E; cb += CHUNK) {
    int ce = g.E - cb; if (ce > CHUNK) ce = CHUNK;
    dim3 gg(ce/16, cdiv(Ncols,64));
    k_gemm_h2w3<<<gg, 32, 0, st>>>(S.h2 + (size_t)cb*32, S.wpack, r.b3, S.Rbuf, ce, Ncols);
    k_pair_msg<<<ce, 64, 0, st>>>(S.Rbuf, g.y1, g.b11, hsrc, g.src, outbuf, di, dop, mo, cb);
  }
}

static void run_gse3(const Graph& g, Scratch& S, const LayerP& L, int cur, int nxt, hipStream_t st) {
  hipMemsetAsync(S.v0, 0, (size_t)g.E*16*4, st);
  hipMemsetAsync(S.v1, 0, (size_t)g.E*48*4, st);
  hipMemsetAsync(S.k0, 0, (size_t)g.E*16*4, st);
  hipMemsetAsync(S.k1, 0, (size_t)g.E*48*4, st);
  // values
  run_rad(g, S, L.v00, 0, 0, 16, S.v0, cur, st);
  if (L.full) run_rad(g, S, L.v10, 1, 0, 16, S.v0, cur, st);
  run_rad(g, S, L.v01, 0, 1, 16, S.v1, cur, st);
  if (L.full) run_rad(g, S, L.v11, 1, 1, 16, S.v1, cur, st);
  // keys
  run_rad(g, S, L.k00, 0, 0, 16, S.k0, cur, st);
  if (L.full) {
    run_rad(g, S, L.k10, 1, 0, 16, S.k0, cur, st);
    run_rad(g, S, L.k01, 0, 1, 16, S.k1, cur, st);
    run_rad(g, S, L.k11, 1, 1, 16, S.k1, cur, st);
  }
  // queries
  k_g1x1<<<cdiv(g.N*16,256), 256, 0, st>>>(L.q0, g.h0[cur], 64, nullptr, 0, S.q0, g.N, 16, 1, 16, 0);
  if (L.full)
    k_g1x1<<<cdiv(g.N*16,256), 256, 0, st>>>(L.q1, g.h1[cur], 64, nullptr, 0, S.q1, g.N, 16, 3, 16, 0);
  // attention (edge softmax over dst segments)
  hipMemsetAsync(S.mbuf, 0, (size_t)g.N*4*4, st);
  hipMemsetAsync(S.ssum, 0, (size_t)g.N*4*4, st);
  float scale = L.full ? 0.125f : 0.25f;  // 1/sqrt(64), 1/sqrt(16)
  int thr = g.E*4;
  k_att_score<<<cdiv(thr,256), 256, 0, st>>>(S.k0, S.k1, S.q0, S.q1, g.dst, S.s, S.mbuf, g.E,
                                             L.full ? 1 : 0, scale);
  k_att_exp<<<cdiv(thr,256), 256, 0, st>>>(S.s, S.mbuf, S.ssum, g.dst, g.E);
  hipMemsetAsync(S.z0, 0, (size_t)g.N*16*4, st);
  hipMemsetAsync(S.z1, 0, (size_t)g.N*48*4, st);
  k_att_scatter<<<cdiv(thr,256), 256, 0, st>>>(S.s, S.ssum, g.dst, S.v0, S.v1, S.z0, S.z1, g.E);
  // projection (concat z with h when deg in degs_in)
  k_g1x1<<<cdiv(g.N*64,256), 256, 0, st>>>(L.proj0, S.z0, 16, g.h0[cur], 64, S.tmp0, g.N, 64, 1, 64, 0);
  if (L.full)
    k_g1x1<<<cdiv(g.N*64,256), 256, 0, st>>>(L.proj1, S.z1, 16, g.h1[cur], 64, S.tmp1, g.N, 64, 3, 64, 0);
  else
    k_g1x1<<<cdiv(g.N*64,256), 256, 0, st>>>(L.proj1, S.z1, 16, nullptr, 0, S.tmp1, g.N, 64, 3, 64, 0);
  // gnorm
  k_gnorm<<<g.N, 64, 0, st>>>(S.tmp0, L.ng0, L.nb0, 1);
  k_gnorm<<<g.N, 64, 0, st>>>(S.tmp1, L.ng1, L.nb1, 3);
  // mid (HA writes channels [0,48) of 64-wide buffer; cross-attn fills the rest)
  k_g1x1<<<cdiv(g.N*g.mid2,256), 256, 0, st>>>(L.mid0, S.tmp0, 64, nullptr, 0, g.h0[nxt],
                                               g.N, g.mid2, 1, 64, 0);
  k_g1x1<<<cdiv(g.N*g.mid2,256), 256, 0, st>>>(L.mid1, S.tmp1, 64, nullptr, 0, g.h1[nxt],
                                               g.N, g.mid2, 3, 64, 0);
}

static void run_final(const Graph& g, Scratch& S, const RadP& f00, const RadP& f10,
                      const float* selfW, float* outp, int cur, hipStream_t st) {
  hipMemsetAsync(S.msg, 0, (size_t)g.E*4, st);
  hipMemsetAsync(S.acc, 0, (size_t)g.N*4, st);
  hipMemsetAsync(S.deg, 0, (size_t)g.N*4, st);
  run_rad(g, S, f00, 0, 0, 1, S.msg, cur, st);
  run_rad(g, S, f10, 1, 0, 1, S.msg, cur, st);
  k_final_scatter<<<cdiv(g.E,256), 256, 0, st>>>(S.msg, selfW, g.h0[cur], g.dst, S.acc, S.deg, g.E);
  k_final_div<<<cdiv(g.N,256), 256, 0, st>>>(S.acc, S.deg, outp, g.N);
}

extern "C" void kernel_launch(void* const* d_in, const int* in_sizes, int n_in,
                              void* d_out, int out_size, void* d_ws, size_t ws_size,
                              hipStream_t stream) {
  (void)in_sizes; (void)n_in; (void)out_size; (void)ws_size;
  const float* f0_ha = (const float*)d_in[0];
  const float* f0_ca = (const float*)d_in[1];
  const float* w_ha  = (const float*)d_in[2];
  const float* w_ca  = (const float*)d_in[3];
  const float* dv_ha = (const float*)d_in[4];
  const float* dv_ca = (const float*)d_in[5];
  const int* src_ha  = (const int*)d_in[6];
  const int* dst_ha  = (const int*)d_in[7];
  const int* src_ca  = (const int*)d_in[8];
  const int* dst_ca  = (const int*)d_in[9];
  int pi = 10;
  GraphP Pca = takeGraph(d_in, pi);   // 'ca' < 'ha' in sorted pytree order
  GraphP Pha = takeGraph(d_in, pi);

  Bump al{(char*)d_ws, 0};
  Graph ha, ca;
  ha.N = NHA; ha.E = EHA; ha.L0 = L0HA; ha.mid2 = 48;
  ha.f0 = f0_ha; ha.src = src_ha; ha.dst = dst_ha;
  ca.N = NCA; ca.E = ECA; ca.L0 = L0CA; ca.mid2 = 64;
  ca.f0 = f0_ca; ca.src = src_ca; ca.dst = dst_ca;
  ha.feat = (float*)bget(al, (size_t)EHA*6*4);
  ha.y1   = (float*)bget(al, (size_t)EHA*3*4);
  ha.b11  = (float*)bget(al, (size_t)EHA*27*4);
  ca.feat = (float*)bget(al, (size_t)ECA*6*4);
  ca.y1   = (float*)bget(al, (size_t)ECA*3*4);
  ca.b11  = (float*)bget(al, (size_t)ECA*27*4);
  for (int j = 0; j < 2; j++) {
    ha.h0[j] = (float*)bget(al, (size_t)NHA*64*4);
    ha.h1[j] = (float*)bget(al, (size_t)NHA*192*4);
    ca.h0[j] = (float*)bget(al, (size_t)NCA*64*4);
    ca.h1[j] = (float*)bget(al, (size_t)NCA*192*4);
  }
  Scratch S;
  S.tmp0 = (float*)bget(al, (size_t)NHA*64*4);
  S.tmp1 = (float*)bget(al, (size_t)NHA*192*4);
  S.embt = (float*)bget(al, (size_t)NHA*L0HA*4);
  S.Rbuf = (float*)bget(al, (size_t)CHUNK*3072*4);
  S.v0   = (float*)bget(al, (size_t)EHA*16*4);
  S.v1   = (float*)bget(al, (size_t)EHA*48*4);
  S.k0   = (float*)bget(al, (size_t)EHA*16*4);
  S.k1   = (float*)bget(al, (size_t)EHA*48*4);
  S.q0   = (float*)bget(al, (size_t)NHA*16*4);
  S.q1   = (float*)bget(al, (size_t)NHA*48*4);
  S.s    = (float*)bget(al, (size_t)EHA*4*4);
  S.ssum = (float*)bget(al, (size_t)NHA*4*4);
  S.z0   = (float*)bget(al, (size_t)NHA*16*4);
  S.z1   = (float*)bget(al, (size_t)NHA*48*4);
  S.msg  = (float*)bget(al, (size_t)EHA*4);
  S.acc  = (float*)bget(al, (size_t)NHA*4);
  S.deg  = (float*)bget(al, (size_t)NHA*4);
  S.mbuf = (unsigned*)bget(al, (size_t)NHA*4*4);
  S.h2   = (_Float16*)bget(al, (size_t)EHA*32*2);
  S.wpack= (_Float16*)bget(al, (size_t)32*3072*2);

  // geometry + features
  k_edge_geom<<<cdiv(EHA,256), 256, 0, stream>>>(w_ha, dv_ha, ha.feat, ha.y1, ha.b11, EHA);
  k_edge_geom<<<cdiv(ECA,256), 256, 0, stream>>>(w_ca, dv_ca, ca.feat, ca.y1, ca.b11, ECA);
  // embedding
  k_linear<<<cdiv(NHA*L0HA,256), 256, 0, stream>>>(f0_ha, Pha.lin1_w, Pha.lin1_b, S.embt, NHA, L0HA, L0HA, 1);
  k_linear<<<cdiv(NHA*64,256),   256, 0, stream>>>(S.embt, Pha.lin2_w, Pha.lin2_b, ha.h0[0], NHA, L0HA, 64, 0);
  k_linear<<<cdiv(NCA*L0CA,256), 256, 0, stream>>>(f0_ca, Pca.lin1_w, Pca.lin1_b, S.embt, NCA, L0CA, L0CA, 1);
  k_linear<<<cdiv(NCA*64,256),   256, 0, stream>>>(S.embt, Pca.lin2_w, Pca.lin2_b, ca.h0[0], NCA, L0CA, 64, 0);

  float* attbase = (float*)d_out + 3072;
  int cur = 0;
  for (int l = 0; l < 4; l++) {
    int nxt = cur ^ 1;
    run_gse3(ha, S, Pha.L[l], cur, nxt, stream);
    run_gse3(ca, S, Pca.L[l], cur, nxt, stream);
    // cross attention: ha queries, ca keys/values; attmaps written to d_out
    float* A0 = attbase + (size_t)(l*2)   * NHA * NCA;
    float* A1 = attbase + (size_t)(l*2+1) * NHA * NCA;
    k_qk<<<dim3(NHA/16, NCA/16), 32, 0, stream>>>(ha.h0[nxt], 64, ca.h0[nxt], 64, A0, NCA, 16, 0.25f);
    k_softmax_row<<<NHA, 256, 0, stream>>>(A0, NCA);
    k_av<<<dim3(NHA/16, 1), 32, 0, stream>>>(A0, NCA, ca.h0[nxt], 64, 16, ha.h0[nxt], 64, 48, 1, 16);
    k_qk<<<dim3(NHA/16, NCA/16), 32, 0, stream>>>(ha.h1[nxt], 192, ca.h1[nxt], 192, A1, NCA, 48, 0.25f);
    k_softmax_row<<<NHA, 256, 0, stream>>>(A1, NCA);
    k_av<<<dim3(NHA/16, 3), 32, 0, stream>>>(A1, NCA, ca.h1[nxt], 192, 48, ha.h1[nxt], 64, 48, 3, 48);
    cur = nxt;
  }
  // final degree-0 conv
  run_final(ha, S, Pha.fin00, Pha.fin10, Pha.fin_self, (float*)d_out,        cur, stream);
  run_final(ca, S, Pca.fin00, Pca.fin10, Pca.fin_self, (float*)d_out + NHA,  cur, stream);
}
